// MultiHeadAttention_69647189672066
// MI455X (gfx1250) — matrix operand order repaired and checked by reading, not compiled, not run on hardware
//
#include <hip/hip_runtime.h>
#include <hip/hip_bf16.h>

#define D_MODEL 1024
#define N_HEADS 16
#define HEAD_DIM 64
#define BATCH 4
#define SEQ 2048
#define BS (BATCH*SEQ)   /* 8192 */

typedef __attribute__((ext_vector_type(16))) _Float16 v16h;
typedef __attribute__((ext_vector_type(8)))  _Float16 v8h;
typedef __attribute__((ext_vector_type(8)))  float    v8f;
typedef __attribute__((ext_vector_type(4)))  float    v4f;
typedef __attribute__((ext_vector_type(4)))  int      v4i;

union FragH { v16h v; v8h h[2]; };

#define AS1 __attribute__((address_space(1)))
#define AS3 __attribute__((address_space(3)))

// ---- CDNA5 async global->LDS path (ASYNCcnt), with safe fallback ----------
#if defined(__has_builtin)
#  if __has_builtin(__builtin_amdgcn_global_load_async_to_lds_b128)
#    define HAVE_ASYNC_LDS 1
#  endif
#  if __has_builtin(__builtin_amdgcn_s_wait_asynccnt)
#    define HAVE_WAIT_ASYNC 1
#  endif
#endif
#ifndef HAVE_ASYNC_LDS
#define HAVE_ASYNC_LDS 0
#endif
#ifndef HAVE_WAIT_ASYNC
#define HAVE_WAIT_ASYNC 0
#endif

// copy 16 bytes (8 f16) global -> LDS, per lane
__device__ __forceinline__ void cp16_to_lds(_Float16* dst, const _Float16* src) {
#if HAVE_ASYNC_LDS
  // signature (probe-derived): (global v4i* src, lds v4i* dst, imm offset, imm cpol)
  __builtin_amdgcn_global_load_async_to_lds_b128(
      (AS1 v4i*)src, (AS3 v4i*)dst, 0, 0);
#else
  *(v8h*)dst = *(const v8h*)src;   // global_load_b128 + ds_store_b128
#endif
}

__device__ __forceinline__ void wait_async_lds() {
#if HAVE_ASYNC_LDS
#  if HAVE_WAIT_ASYNC
  __builtin_amdgcn_s_wait_asynccnt(0);
#  else
  asm volatile("s_wait_asynccnt 0" ::: "memory");
#  endif
#endif
}

__device__ __forceinline__ v8f vzero8() {
  v8f z = {0.f,0.f,0.f,0.f,0.f,0.f,0.f,0.f};
  return z;
}

__device__ __forceinline__ v8f wmma_f16(const FragH& a, const FragH& b, v8f c) {
  // D = A(16x32 f16) * B(32x16 f16) + C(16x16 f32)
  return __builtin_amdgcn_wmma_f32_16x16x32_f16(
      /*neg_a=*/false, a.v, /*neg_b=*/false, b.v,
      /*c_mod=*/(short)0, c, /*reuse_a=*/false, /*reuse_b=*/false);
}

// ---------------------------------------------------------------------------
// fp32 -> f16 conversion (one-time; bandwidth-trivial vs 23.3 TB/s)
// ---------------------------------------------------------------------------
__global__ __launch_bounds__(256)
void cvt_f32_f16(const float* __restrict__ in, _Float16* __restrict__ out, int n) {
  int i = (blockIdx.x * 256 + threadIdx.x) * 8;
  if (i >= n) return;
  v4f a = *(const v4f*)(in + i);
  v4f b = *(const v4f*)(in + i + 4);
  v8h o;
#pragma unroll
  for (int j = 0; j < 4; ++j) { o[j] = (_Float16)a[j]; o[4 + j] = (_Float16)b[j]; }
  *(v8h*)(out + i) = o;
}

// ---------------------------------------------------------------------------
// GEMM: out[M,1024] = A[M,1024] * W[1024,1024]^T   (out[m,n] = sum_k A[m,k]W[n,k])
// 256 threads = 8 waves, wave grid 4(M) x 2(N); wave tile 16x64; WG tile 64x128.
// B tile (32k x 128n) staged in double-buffered LDS via async copies: shared by
// the 4 M-waves (4x less L2 traffic), next k-step streamed during current WMMAs.
// MODE 0: store f16 row-major [M,1024]             (Q, K projections)
// MODE 1: store f16 transposed per head [B,H,64,S] (V projection -> attn B-operand)
// MODE 2: store fp32 row-major + bias              (output projection)
// ---------------------------------------------------------------------------
template<int MODE>
__global__ __launch_bounds__(256)
void gemm_f16(const _Float16* __restrict__ A, const _Float16* __restrict__ W,
              _Float16* __restrict__ outh, float* __restrict__ outf,
              const float* __restrict__ bias) {
  __shared__ _Float16 Bt[2][128][40];   // [buf][n_local][k], pitch 80B (16B-aligned)

  const int lane = threadIdx.x & 31;
  const int wv   = threadIdx.x >> 5;
  const int l15  = lane & 15;
  const int hf   = lane >> 4;
  const int wm   = wv & 3;
  const int wn   = wv >> 2;

  const int m0    = blockIdx.x * 64 + wm * 16;   // wave M base
  const int n0blk = blockIdx.y * 128;            // WG N base
  const int n0    = n0blk + wn * 64;             // wave N base (64 wide)

  const _Float16* Arow = A + (size_t)(m0 + l15) * D_MODEL;

  // staging assignment: thread t covers B row n = t>>1, k-chunk (t&1)*16
  const int tn = threadIdx.x >> 1;
  const int tc = (threadIdx.x & 1) * 16;
  const _Float16* Wst = W + (size_t)(n0blk + tn) * D_MODEL + tc;

  v8f acc[4];
#pragma unroll
  for (int i = 0; i < 4; ++i) acc[i] = vzero8();

  // prologue: stage k-step 0 into buffer 0
  cp16_to_lds(&Bt[0][tn][tc],     Wst);
  cp16_to_lds(&Bt[0][tn][tc + 8], Wst + 8);

  for (int it = 0; it < D_MODEL / 32; ++it) {
    wait_async_lds();
    __syncthreads();                       // buffer (it&1) fully landed, old reads drained
    if (it + 1 < D_MODEL / 32) {           // stream next k-step into the other buffer
      const _Float16* Wnext = Wst + (it + 1) * 32;
      cp16_to_lds(&Bt[(it + 1) & 1][tn][tc],     Wnext);
      cp16_to_lds(&Bt[(it + 1) & 1][tn][tc + 8], Wnext + 8);
    }
    const int k0  = it * 32;
    const int buf = it & 1;

    FragH a;   // A 16x32: lane(M=l15,hf): K = hf*16 + e (contiguous)
    a.h[0] = *(const v8h*)(Arow + k0 + hf * 16);
    a.h[1] = *(const v8h*)(Arow + k0 + hf * 16 + 8);
    __builtin_prefetch(Arow + k0 + 64, 0, 1);   // global_prefetch_b8
#pragma unroll
    for (int nt = 0; nt < 4; ++nt) {
      // B 32x16: lane(N=l15,hf): K = hf*16 + e -> 32B contiguous in LDS row
      FragH b;
      b.h[0] = *(const v8h*)(&Bt[buf][wn * 64 + nt * 16 + l15][hf * 16]);
      b.h[1] = *(const v8h*)(&Bt[buf][wn * 64 + nt * 16 + l15][hf * 16 + 8]);
      acc[nt] = wmma_f16(a, b, acc[nt]);
    }
  }

  if (MODE == 0) {
#pragma unroll
    for (int nt = 0; nt < 4; ++nt) {
      const int col = n0 + nt * 16 + l15;
#pragma unroll
      for (int r = 0; r < 8; ++r) {
        const int row = m0 + 8 * hf + r;      // C layout: M = r + 8*hf
        outh[(size_t)row * D_MODEL + col] = (_Float16)acc[nt][r];
      }
    }
  } else if (MODE == 1) {
    const int b     = m0 / SEQ;               // 64-row tiles never straddle batch
    const int sbase = (m0 % SEQ) + 8 * hf;    // 8 consecutive s per lane
#pragma unroll
    for (int nt = 0; nt < 4; ++nt) {
      const int col = n0 + nt * 16 + l15;
      const int h   = col >> 6;
      const int d   = col & 63;
      v8h pk;
#pragma unroll
      for (int r = 0; r < 8; ++r) pk[r] = (_Float16)acc[nt][r];
      *(v8h*)(outh + ((size_t)((b * N_HEADS + h) * HEAD_DIM + d)) * SEQ + sbase) = pk;
    }
  } else {
#pragma unroll
    for (int nt = 0; nt < 4; ++nt) {
      const int col = n0 + nt * 16 + l15;
      const float bv = bias[col];
#pragma unroll
      for (int r = 0; r < 8; ++r) {
        const int row = m0 + 8 * hf + r;
        outf[(size_t)row * D_MODEL + col] = acc[nt][r] + bv;
      }
    }
  }
}

// ---------------------------------------------------------------------------
// RoPE on Q and K (f16 in/out, fp32 math). Row = b*S+s in [BS,1024] layout.
// Pair (d, d+32) within each head; angle = s * 10000^(-d/32), same for both.
// ---------------------------------------------------------------------------
__global__ __launch_bounds__(256)
void rope_kernel(_Float16* __restrict__ Q, _Float16* __restrict__ K) {
  const int idx = blockIdx.x * 256 + threadIdx.x;   // BS * 16 * 32 threads
  const int row = idx >> 9;
  const int rem = idx & 511;
  const int h = rem >> 5;
  const int d = rem & 31;
  const int s = row & (SEQ - 1);

  const float inv = __powf(10000.0f, -(float)d * (1.0f / 32.0f));
  float sn, cs;
  __sincosf((float)s * inv, &sn, &cs);

  const size_t base = (size_t)row * D_MODEL + h * HEAD_DIM + d;
  {
    float x1 = (float)Q[base], x2 = (float)Q[base + 32];
    Q[base]      = (_Float16)(x1 * cs - x2 * sn);
    Q[base + 32] = (_Float16)(x2 * cs + x1 * sn);
  }
  {
    float x1 = (float)K[base], x2 = (float)K[base + 32];
    K[base]      = (_Float16)(x1 * cs - x2 * sn);
    K[base + 32] = (_Float16)(x2 * cs + x1 * sn);
  }
}

// ---------------------------------------------------------------------------
// Flash attention. Grid (S/64, H, B), 128 threads = 4 waves.
// Wave w owns q rows [q0+16w, q0+16w+16). Per 32-kv step:
//   - K tile [32 x 64] and Vt tile [64 x 32] double-buffered in LDS, streamed
//     with async copies while the previous tile is being consumed
//   - scores 16x32 via 4 WMMA, online softmax fp32 (shfl_xor over 16-lane half)
//   - P -> per-wave LDS (C-layout -> A-layout reshape) -> 4 WMMA against V
// ---------------------------------------------------------------------------
__global__ __launch_bounds__(128)
void attn_kernel(const _Float16* __restrict__ Q, const _Float16* __restrict__ K,
                 const _Float16* __restrict__ Vt, _Float16* __restrict__ AO) {
  __shared__ _Float16 Kt[2][32][72];     // padded pitch 144B (16B-aligned rows)
  __shared__ _Float16 Vs[2][64][40];     // Vt tile: row = head-dim d, col = kv
  __shared__ _Float16 Pw[4][16][40];     // per-wave P scratch 16x32 (padded)

  const int t    = threadIdx.x;
  const int lane = t & 31;
  const int wv   = t >> 5;
  const int l15  = lane & 15;
  const int hf   = lane >> 4;
  const int b    = blockIdx.z;
  const int h    = blockIdx.y;
  const int q0   = blockIdx.x * 64 + wv * 16;

  // Q fragments (K-dim = head dim 64 -> two 16x32 A fragments), resident all loop
  const _Float16* Qrow = Q + (size_t)(b * SEQ + q0 + l15) * D_MODEL + h * HEAD_DIM;
  FragH qa0, qa1;
  qa0.h[0] = *(const v8h*)(Qrow + hf * 16);
  qa0.h[1] = *(const v8h*)(Qrow + hf * 16 + 8);
  qa1.h[0] = *(const v8h*)(Qrow + 32 + hf * 16);
  qa1.h[1] = *(const v8h*)(Qrow + 32 + hf * 16 + 8);

  v8f o0 = vzero8(), o1 = vzero8(), o2 = vzero8(), o3 = vzero8();
  float mrun[8], lrun[8];
#pragma unroll
  for (int r = 0; r < 8; ++r) { mrun[r] = -1e30f; lrun[r] = 0.f; }

  const _Float16* Kbase = K + (size_t)(b * SEQ) * D_MODEL + h * HEAD_DIM;
  const _Float16* Vbase = Vt + (size_t)((b * N_HEADS + h) * HEAD_DIM) * SEQ;

  // staging assignment (per thread): K row k=t>>2 chunk (t&3)*16; V row d=t>>1
  const int sk = t >> 2, sc = (t & 3) * 16;
  const int sd = t >> 1, sc2 = (t & 1) * 16;

  auto stageKV = [&](int buf, int kv0) {
    const _Float16* ks = Kbase + (size_t)(kv0 + sk) * D_MODEL + sc;
    cp16_to_lds(&Kt[buf][sk][sc],     ks);
    cp16_to_lds(&Kt[buf][sk][sc + 8], ks + 8);
    const _Float16* vs = Vbase + (size_t)sd * SEQ + kv0 + sc2;
    cp16_to_lds(&Vs[buf][sd][sc2],     vs);
    cp16_to_lds(&Vs[buf][sd][sc2 + 8], vs + 8);
  };

  stageKV(0, 0);

  for (int it = 0; it < SEQ / 32; ++it) {
    wait_async_lds();
    __syncthreads();                        // tile landed; old-buffer reads drained
    if (it + 1 < SEQ / 32) stageKV((it + 1) & 1, (it + 1) * 32);
    const int buf = it & 1;

    // scores: two 16x16 tiles (kv cols 0-15 and 16-31), K-dim 64 -> 2 WMMA each.
    // B[d][n] = K[kv0+n][d]: lane(n,hf) reads 32B of Kt row n at d-offset hf*16.
    v8f s0 = vzero8(), s1 = vzero8();
    FragH bk;
    bk.h[0] = *(const v8h*)(&Kt[buf][l15][hf * 16]);
    bk.h[1] = *(const v8h*)(&Kt[buf][l15][hf * 16 + 8]);
    s0 = wmma_f16(qa0, bk, s0);
    bk.h[0] = *(const v8h*)(&Kt[buf][l15][32 + hf * 16]);
    bk.h[1] = *(const v8h*)(&Kt[buf][l15][32 + hf * 16 + 8]);
    s0 = wmma_f16(qa1, bk, s0);
    bk.h[0] = *(const v8h*)(&Kt[buf][16 + l15][hf * 16]);
    bk.h[1] = *(const v8h*)(&Kt[buf][16 + l15][hf * 16 + 8]);
    s1 = wmma_f16(qa0, bk, s1);
    bk.h[0] = *(const v8h*)(&Kt[buf][16 + l15][32 + hf * 16]);
    bk.h[1] = *(const v8h*)(&Kt[buf][16 + l15][32 + hf * 16 + 8]);
    s1 = wmma_f16(qa1, bk, s1);

    // online softmax: row m = r + 8*hf spread over 16 lanes of the half;
    // xor masks 1,2,4,8 reduce within the half (never mix rows).
#pragma unroll
    for (int r = 0; r < 8; ++r) {
      float v0 = s0[r] * 0.125f, v1 = s1[r] * 0.125f;  // 1/sqrt(64)
      float mx = fmaxf(v0, v1);
      mx = fmaxf(mx, __shfl_xor(mx, 1, 32));
      mx = fmaxf(mx, __shfl_xor(mx, 2, 32));
      mx = fmaxf(mx, __shfl_xor(mx, 4, 32));
      mx = fmaxf(mx, __shfl_xor(mx, 8, 32));
      const float mn    = fmaxf(mrun[r], mx);
      const float alpha = __expf(mrun[r] - mn);
      const float p0 = __expf(v0 - mn);
      const float p1 = __expf(v1 - mn);
      float rs = p0 + p1;
      rs += __shfl_xor(rs, 1, 32);
      rs += __shfl_xor(rs, 2, 32);
      rs += __shfl_xor(rs, 4, 32);
      rs += __shfl_xor(rs, 8, 32);
      lrun[r] = lrun[r] * alpha + rs;
      mrun[r] = mn;
      o0[r] *= alpha; o1[r] *= alpha; o2[r] *= alpha; o3[r] *= alpha;
      Pw[wv][8 * hf + r][l15]      = (_Float16)p0;   // C-layout -> LDS
      Pw[wv][8 * hf + r][16 + l15] = (_Float16)p1;
    }
    __syncthreads();   // P visible (also keeps barrier count uniform)

    // re-load P in A-matrix layout (16x32)
    FragH pa;
    pa.h[0] = *(const v8h*)(&Pw[wv][l15][hf * 16]);
    pa.h[1] = *(const v8h*)(&Pw[wv][l15][hf * 16 + 8]);

    // P(16x32) x V(32x64): B[k][n] = V[kv0+k][d0+n] = Vs[d0+n][k] (contiguous)
    FragH bv;
    bv.h[0] = *(const v8h*)(&Vs[buf][l15][hf * 16]);
    bv.h[1] = *(const v8h*)(&Vs[buf][l15][hf * 16 + 8]);
    o0 = wmma_f16(pa, bv, o0);
    bv.h[0] = *(const v8h*)(&Vs[buf][16 + l15][hf * 16]);
    bv.h[1] = *(const v8h*)(&Vs[buf][16 + l15][hf * 16 + 8]);
    o1 = wmma_f16(pa, bv, o1);
    bv.h[0] = *(const v8h*)(&Vs[buf][32 + l15][hf * 16]);
    bv.h[1] = *(const v8h*)(&Vs[buf][32 + l15][hf * 16 + 8]);
    o2 = wmma_f16(pa, bv, o2);
    bv.h[0] = *(const v8h*)(&Vs[buf][48 + l15][hf * 16]);
    bv.h[1] = *(const v8h*)(&Vs[buf][48 + l15][hf * 16 + 8]);
    o3 = wmma_f16(pa, bv, o3);
  }

  // normalize and store attention output as f16 [BS, 1024] for the final GEMM
#pragma unroll
  for (int r = 0; r < 8; ++r) {
    const float il = 1.0f / lrun[r];
    const size_t row = (size_t)(b * SEQ + q0 + 8 * hf + r) * D_MODEL + h * HEAD_DIM;
    AO[row + l15]      = (_Float16)(o0[r] * il);
    AO[row + 16 + l15] = (_Float16)(o1[r] * il);
    AO[row + 32 + l15] = (_Float16)(o2[r] * il);
    AO[row + 48 + l15] = (_Float16)(o3[r] * il);
  }
}

// ---------------------------------------------------------------------------
extern "C" void kernel_launch(void* const* d_in, const int* in_sizes, int n_in,
                              void* d_out, int out_size, void* d_ws, size_t ws_size,
                              hipStream_t stream) {
  const float* X  = (const float*)d_in[0];
  const float* Wq = (const float*)d_in[1];
  const float* Wk = (const float*)d_in[2];
  const float* Wv = (const float*)d_in[3];
  const float* Wo = (const float*)d_in[4];
  const float* bo = (const float*)d_in[5];
  float* out = (float*)d_out;

  char* p = (char*)d_ws;
  auto carve = [&p](size_t bytes) {
    char* r = p;
    p += (bytes + 255) & ~(size_t)255;
    return r;
  };
  const size_t actB = (size_t)BS * D_MODEL * sizeof(_Float16);       // 16 MB
  const size_t wB   = (size_t)D_MODEL * D_MODEL * sizeof(_Float16);  // 2 MB
  _Float16* Xb  = (_Float16*)carve(actB);
  _Float16* Wqb = (_Float16*)carve(wB);
  _Float16* Wkb = (_Float16*)carve(wB);
  _Float16* Wvb = (_Float16*)carve(wB);
  _Float16* Wob = (_Float16*)carve(wB);
  _Float16* Qb  = (_Float16*)carve(actB);
  _Float16* Kb  = (_Float16*)carve(actB);
  _Float16* Vb  = (_Float16*)carve(actB);   // [B,H,64,S]
  _Float16* AOb = (_Float16*)carve(actB);

  // 1) fp32 -> f16 (X and the four weights)
  cvt_f32_f16<<<(BS * D_MODEL) / 2048, 256, 0, stream>>>(X, Xb, BS * D_MODEL);
  cvt_f32_f16<<<(D_MODEL * D_MODEL) / 2048, 256, 0, stream>>>(Wq, Wqb, D_MODEL * D_MODEL);
  cvt_f32_f16<<<(D_MODEL * D_MODEL) / 2048, 256, 0, stream>>>(Wk, Wkb, D_MODEL * D_MODEL);
  cvt_f32_f16<<<(D_MODEL * D_MODEL) / 2048, 256, 0, stream>>>(Wv, Wvb, D_MODEL * D_MODEL);
  cvt_f32_f16<<<(D_MODEL * D_MODEL) / 2048, 256, 0, stream>>>(Wo, Wob, D_MODEL * D_MODEL);

  // 2) Q/K/V projections (V written pre-transposed per head)
  dim3 gg(BS / 64, D_MODEL / 128);
  gemm_f16<0><<<gg, 256, 0, stream>>>(Xb, Wqb, Qb, nullptr, nullptr);
  gemm_f16<0><<<gg, 256, 0, stream>>>(Xb, Wkb, Kb, nullptr, nullptr);
  gemm_f16<1><<<gg, 256, 0, stream>>>(Xb, Wvb, Vb, nullptr, nullptr);

  // 3) RoPE on Q and K
  rope_kernel<<<(BS * N_HEADS * 32) / 256, 256, 0, stream>>>(Qb, Kb);

  // 4) flash attention per (b, h, 64-row q block)
  attn_kernel<<<dim3(SEQ / 64, N_HEADS, BATCH), 128, 0, stream>>>(Qb, Kb, Vb, AOb);

  // 5) output projection (fp32 out + bias)
  gemm_f16<2><<<gg, 256, 0, stream>>>(AOb, Wob, nullptr, out, bo);
}
